// RoutingMamba_84602265796866
// MI455X (gfx1250) — compile-verified, matrix operand order checked by reading, not run on hardware
//
#include <hip/hip_runtime.h>

// ---------------------------------------------------------------------------
// RoutingMamba for MI455X (gfx1250, wave32).
// All-expert dense compute is GEMM-dominated (~0.83 TFLOP) and compute-bound
// (HBM floor ~11us @ 23.3TB/s), so all heavy math goes through
// v_wmma_f32_16x16x32_bf16 (f32 accumulate). Tile staging uses the CDNA5
// async path: global_load_async_to_lds_b128 + ASYNCcnt, double-buffered LDS.
// ---------------------------------------------------------------------------

typedef __attribute__((ext_vector_type(16))) __bf16 v16bf;
typedef __attribute__((ext_vector_type(8)))  float  v8f;

#define B_    4
#define S_    2048
#define D_    1024
#define DI_   2048          // D_INNER
#define E_    8
#define RH_   128
#define NTOK  (B_ * S_)     // 8192

__device__ __forceinline__ unsigned short f2bf(float f) {
  unsigned int u = __float_as_uint(f);
  u += 0x7FFFu + ((u >> 16) & 1u);           // round to nearest even
  return (unsigned short)(u >> 16);
}
__device__ __forceinline__ float bf2f(unsigned short h) {
  return __uint_as_float(((unsigned int)h) << 16);
}
__device__ __forceinline__ float silu_f(float v) {
  return v * (1.0f / (1.0f + __expf(-v)));
}

// ---------------------------------------------------------------------------
// fp32 -> bf16 bulk convert (grid-stride)
// ---------------------------------------------------------------------------
__global__ void cvt_f32_bf16(const float* __restrict__ src,
                             unsigned short* __restrict__ dst, int n) {
  int i = blockIdx.x * blockDim.x + threadIdx.x;
  int stride = gridDim.x * blockDim.x;
  for (; i < n; i += stride) dst[i] = f2bf(src[i]);
}

// ---------------------------------------------------------------------------
// Router: one wave32 per token. rh = silu(h @ w1^T) (128), logits = rh @ w2^T,
// softmax, top-2, renormalize -> comb[token][E_].
// ---------------------------------------------------------------------------
__global__ __launch_bounds__(256)
void router_kernel(const float* __restrict__ h,
                   const float* __restrict__ w1,   // [RH_, D_]
                   const float* __restrict__ w2,   // [E_, RH_]
                   float* __restrict__ comb) {     // [NTOK, E_]
  __shared__ float sh[8][RH_];
  __shared__ float slog[8][E_];
  const int wave = threadIdx.x >> 5;
  const int lane = threadIdx.x & 31;
  const int token = blockIdx.x * 8 + wave;
  const float* hrow = h + (size_t)token * D_;

  float acc[4] = {0.f, 0.f, 0.f, 0.f};
  for (int k = 0; k < D_; ++k) {
    const float hk = hrow[k];
#pragma unroll
    for (int p = 0; p < 4; ++p)
      acc[p] += hk * w1[(lane + 32 * p) * D_ + k];
  }
#pragma unroll
  for (int p = 0; p < 4; ++p) sh[wave][lane + 32 * p] = silu_f(acc[p]);
  __syncthreads();

  if (lane < E_) {
    float lg = 0.f;
    for (int j = 0; j < RH_; ++j) lg += sh[wave][j] * w2[lane * RH_ + j];
    slog[wave][lane] = lg;
  }
  __syncthreads();

  if (lane == 0) {
    float mx = -1e30f;
    for (int e = 0; e < E_; ++e) mx = fmaxf(mx, slog[wave][e]);
    float pr[E_], den = 0.f;
    for (int e = 0; e < E_; ++e) { pr[e] = __expf(slog[wave][e] - mx); den += pr[e]; }
    const float rden = 1.0f / den;
    for (int e = 0; e < E_; ++e) pr[e] *= rden;
    int i0 = 0;
    for (int e = 1; e < E_; ++e) if (pr[e] > pr[i0]) i0 = e;
    int i1 = (i0 == 0) ? 1 : 0;
    for (int e = 0; e < E_; ++e) if (e != i0 && pr[e] > pr[i1]) i1 = e;
    const float rw = 1.0f / (pr[i0] + pr[i1]);
    for (int e = 0; e < E_; ++e) comb[(size_t)token * E_ + e] = 0.0f;
    comb[(size_t)token * E_ + i0] = pr[i0] * rw;
    comb[(size_t)token * E_ + i1] = pr[i1] * rw;
  }
}

// ---------------------------------------------------------------------------
// bf16 WMMA GEMM: C[M,N] = A[M,K] * W[N,K]^T   (torch Linear layout)
// 128x128x32 block tile, 256 threads = 8 waves; each wave owns a 32x64 tile
// as 2x4 v_wmma_f32_16x16x32_bf16 accumulators.
// Tiles staged via GLOBAL_LOAD_ASYNC_TO_LDS_B128 (ASYNCcnt) into
// double-buffered LDS so copies overlap with the WMMA pipe.
// EPI==0: store bf16 C.     EPI==1: out[m,n] (+)= comb[m,expert] * C[m,n].
// ---------------------------------------------------------------------------
template <int EPI>
__global__ __launch_bounds__(256)
void gemm_bf16_wmma(const unsigned short* __restrict__ A,
                    const unsigned short* __restrict__ W,
                    int M, int N, int K,
                    unsigned short* __restrict__ Cbf,
                    float* __restrict__ Cf,
                    const float* __restrict__ comb,
                    int expert, int accumulate) {
  constexpr int BM = 128, BN = 128, BK = 32;
  __shared__ unsigned short lA[2][BM * BK];   // 2 x 8 KB
  __shared__ unsigned short lB[2][BN * BK];   // 2 x 8 KB

  const int tid  = threadIdx.x;
  const int lane = tid & 31;
  const int wave = tid >> 5;
  const int wm = wave & 3;    // 4 wave-rows  * 32 M
  const int wn = wave >> 2;   // 2 wave-cols  * 64 N
  const int bm = blockIdx.y * BM;
  const int bn = blockIdx.x * BN;

  const int ldrow  = tid >> 1;   // 0..127
  const int ldhalf = tid & 1;    // 16-ushort (32 B) chunk within a 32-wide row

  v8f acc[2][4] = {};

  const int rsel  = lane & 15;
  const int khalf = (lane >> 4) & 1;

  // Per-thread async staging: 32 B of A + 32 B of W per tile as 2+2 B128
  // async copies, LDS written directly (no VGPR round trip). INST_OFFSET is
  // added to BOTH the global and LDS address per the ISA, so offset:16 covers
  // the second 16-byte chunk on both sides.
  auto issue_tile = [&](int k0, int buf) {
    const unsigned la =
        (unsigned)(unsigned long long)&lA[buf][ldrow * BK + ldhalf * 16];
    const unsigned long long ga = (unsigned long long)
        (A + (size_t)(bm + ldrow) * K + k0 + ldhalf * 16);
    asm volatile("global_load_async_to_lds_b128 %0, %1, off\n\t"
                 "global_load_async_to_lds_b128 %0, %1, off offset:16"
                 :: "v"(la), "v"(ga) : "memory");
    const unsigned lb =
        (unsigned)(unsigned long long)&lB[buf][ldrow * BK + ldhalf * 16];
    const unsigned long long gb = (unsigned long long)
        (W + (size_t)(bn + ldrow) * K + k0 + ldhalf * 16);
    asm volatile("global_load_async_to_lds_b128 %0, %1, off\n\t"
                 "global_load_async_to_lds_b128 %0, %1, off offset:16"
                 :: "v"(lb), "v"(gb) : "memory");
  };

  issue_tile(0, 0);

  int buf = 0;
  for (int k0 = 0; k0 < K; k0 += BK, buf ^= 1) {
    if (k0 + BK < K) {
      issue_tile(k0 + BK, buf ^ 1);          // prefetch next slab into other buf
      // 8 async loads outstanding; ASYNCcnt decrements in issue order, so
      // <=4 means the CURRENT buffer's 4 copies have landed.
      asm volatile("s_wait_asynccnt 0x4" ::: "memory");
    } else {
      asm volatile("s_wait_asynccnt 0x0" ::: "memory");
    }
    if (k0 + 2 * BK < K) {                   // warm L2 two slabs ahead
      __builtin_prefetch(A + (size_t)(bm + ldrow) * K + k0 + 2 * BK + ldhalf * 16, 0, 1);
      __builtin_prefetch(W + (size_t)(bn + ldrow) * K + k0 + 2 * BK + ldhalf * 16, 0, 1);
    }
    __syncthreads();                         // tile published to all waves

    v16bf afrag[2];
#pragma unroll
    for (int i = 0; i < 2; ++i)
      afrag[i] = *(const v16bf*)(&lA[buf][(wm * 32 + i * 16 + rsel) * BK + khalf * 16]);
    v16bf bfrag[4];
#pragma unroll
    for (int j = 0; j < 4; ++j)
      bfrag[j] = *(const v16bf*)(&lB[buf][(wn * 64 + j * 16 + rsel) * BK + khalf * 16]);

#pragma unroll
    for (int i = 0; i < 2; ++i)
#pragma unroll
      for (int j = 0; j < 4; ++j)
        acc[i][j] = __builtin_amdgcn_wmma_f32_16x16x32_bf16(
            /*neg_a=*/false, afrag[i], /*neg_b=*/false, bfrag[j],
            /*c_mod=*/(short)0, acc[i][j], /*reuse_a=*/false, /*reuse_b=*/false);

    __syncthreads();                         // everyone done reading buf
  }

  // Epilogue. C/D f32 layout: VGPR r, lanes 0-15 -> M=r, lanes 16-31 -> M=8+r.
  const int rlo    = lane & 15;
  const int rhiadd = (lane >> 4) * 8;
#pragma unroll
  for (int i = 0; i < 2; ++i) {
#pragma unroll
    for (int j = 0; j < 4; ++j) {
#pragma unroll
      for (int r = 0; r < 8; ++r) {
        const int m = bm + wm * 32 + i * 16 + rhiadd + r;
        const int n = bn + wn * 64 + j * 16 + rlo;
        const float v = acc[i][j][r];
        if (EPI == 0) {
          Cbf[(size_t)m * N + n] = f2bf(v);
        } else {
          const float o = v * comb[(size_t)m * E_ + expert];
          const size_t idx = (size_t)m * N + n;
          if (accumulate) Cf[idx] += o; else Cf[idx] = o;
        }
      }
    }
  }
}

// ---------------------------------------------------------------------------
// Causal depthwise conv (k=4, left pad 3) + bias + silu(x)*silu(z) gating.
// zx: [NTOK, 2*DI_] bf16 (z cols 0..DI_-1, x cols DI_..2*DI_-1)
// ---------------------------------------------------------------------------
__global__ __launch_bounds__(256)
void conv_gate_kernel(const unsigned short* __restrict__ zx,
                      const float* __restrict__ cw,   // [DI_,4] this expert
                      const float* __restrict__ cb,   // [DI_]
                      unsigned short* __restrict__ xact) {
  const int token = blockIdx.y;
  const int c = blockIdx.x * blockDim.x + threadIdx.x;
  const int s = token & (S_ - 1);
  const size_t rowstride = 2 * DI_;

  float accv = cb[c];
#pragma unroll
  for (int t = 0; t < 4; ++t) {
    const int dt = 3 - t;                    // conv_w tap t touches x[s-(3-t)]
    float xv = 0.0f;
    if (s - dt >= 0)
      xv = bf2f(zx[(size_t)(token - dt) * rowstride + DI_ + c]);
    accv += cw[c * 4 + t] * xv;
  }
  const float z = bf2f(zx[(size_t)token * rowstride + c]);
  xact[(size_t)token * DI_ + c] = f2bf(silu_f(accv) * silu_f(z));
}

// ---------------------------------------------------------------------------
// Host launch
// ---------------------------------------------------------------------------
extern "C" void kernel_launch(void* const* d_in, const int* in_sizes, int n_in,
                              void* d_out, int out_size, void* d_ws, size_t ws_size,
                              hipStream_t stream) {
  const float* h      = (const float*)d_in[0];   // [B,S,D]
  const float* in_w   = (const float*)d_in[1];   // [E, 2*DI, D]
  const float* conv_w = (const float*)d_in[2];   // [E, DI, 1, 4]
  const float* conv_b = (const float*)d_in[3];   // [E, DI]
  const float* out_w  = (const float*)d_in[4];   // [E, D, DI]
  const float* rw1    = (const float*)d_in[5];   // [RH, D]
  const float* rw2    = (const float*)d_in[6];   // [E, RH]
  float* out = (float*)d_out;                    // [B,S,D]

  // ---- workspace partition (~218 MB) ----
  char* ws = (char*)d_ws;
  size_t off = 0;
  auto alloc = [&](size_t bytes) -> void* {
    void* p = ws + off;
    off += (bytes + 255) & ~(size_t)255;
    return p;
  };
  unsigned short* h_bf    = (unsigned short*)alloc((size_t)NTOK * D_ * 2);
  unsigned short* win_bf  = (unsigned short*)alloc((size_t)E_ * 2 * DI_ * D_ * 2);
  unsigned short* wout_bf = (unsigned short*)alloc((size_t)E_ * D_ * DI_ * 2);
  unsigned short* zx_bf   = (unsigned short*)alloc((size_t)NTOK * 2 * DI_ * 2);
  unsigned short* xact_bf = (unsigned short*)alloc((size_t)NTOK * DI_ * 2);
  float*          comb    = (float*)alloc((size_t)NTOK * E_ * 4);

  // ---- precision conversion (bf16 feeds the matrix cores) ----
  cvt_f32_bf16<<<2048, 256, 0, stream>>>(h, h_bf, NTOK * D_);
  cvt_f32_bf16<<<4096, 256, 0, stream>>>(in_w, win_bf, E_ * 2 * DI_ * D_);
  cvt_f32_bf16<<<4096, 256, 0, stream>>>(out_w, wout_bf, E_ * D_ * DI_);

  // ---- routing ----
  router_kernel<<<NTOK / 8, 256, 0, stream>>>(h, rw1, rw2, comb);

  // ---- per-expert dense pipeline ----
  for (int e = 0; e < E_; ++e) {
    // zx = h @ in_w[e]^T : [8192, 4096]
    gemm_bf16_wmma<0><<<dim3((2 * DI_) / 128, NTOK / 128), 256, 0, stream>>>(
        h_bf, win_bf + (size_t)e * 2 * DI_ * D_, NTOK, 2 * DI_, D_,
        zx_bf, nullptr, nullptr, 0, 0);

    // causal conv + gate : xact [8192, 2048]
    conv_gate_kernel<<<dim3(DI_ / 256, NTOK), 256, 0, stream>>>(
        zx_bf, conv_w + (size_t)e * DI_ * 4, conv_b + (size_t)e * DI_, xact_bf);

    // out (+)= comb[:,e] * (xact @ out_w[e]^T) : [8192, 1024]
    gemm_bf16_wmma<1><<<dim3(D_ / 128, NTOK / 128), 256, 0, stream>>>(
        xact_bf, wout_bf + (size_t)e * D_ * DI_, NTOK, D_, DI_,
        nullptr, out, comb, e, (e > 0) ? 1 : 0);
  }
}
